// DeepseekMoELayer_76596446757257
// MI455X (gfx1250) — matrix operand order repaired and checked by reading, not compiled, not run on hardware
//
#include <hip/hip_runtime.h>
#include <math.h>

// ---------------------------------------------------------------------------
// MoE layer for MI455X (gfx1250): bf16 WMMA GEMMs, wave32.
// A tile: per-lane async global->LDS (ASYNCcnt). B tile: TDM descriptor
// (TENSORcnt) with LDS pad insertion. Double-buffered, last-iter peeled.
// ---------------------------------------------------------------------------

typedef __bf16 bf16;
typedef __attribute__((ext_vector_type(16))) __bf16 v16bf;
typedef __attribute__((ext_vector_type(8)))  float  v8f;
typedef __attribute__((ext_vector_type(4)))  int    v4i;
typedef __attribute__((ext_vector_type(4)))  unsigned int u32x4;
typedef __attribute__((ext_vector_type(8)))  int    i32x8;
typedef __attribute__((ext_vector_type(4)))  int    i32x4;

#define N_EMBD 1024
#define HIDDEN 4096
#define N_EXP  8
#define NTOK   8192

// GEMM tiling
#define BM 128
#define BN 128
#define BK 32
#define LDA_P (BK + 8)   // bf16 pitch, 80B rows (16B aligned)
#define LDB_P (BN + 8)   // bf16 pitch, 272B rows = 256B data + 16B pad

#if __has_builtin(__builtin_amdgcn_global_load_async_to_lds_b128)
#define HAS_ASYNC_LDS 1
#else
#define HAS_ASYNC_LDS 0
#endif

#if __has_builtin(__builtin_amdgcn_tensor_load_to_lds) && \
    __has_builtin(__builtin_amdgcn_s_wait_tensorcnt)
#define HAS_TDM 1
#else
#define HAS_TDM 0
#endif

#if HAS_ASYNC_LDS
#define WAIT_ASYNCCNT(n) __builtin_amdgcn_s_wait_asynccnt(n)
#else
#define WAIT_ASYNCCNT(n)
#endif
#if HAS_TDM
#define WAIT_TENSORCNT(n) __builtin_amdgcn_s_wait_tensorcnt(n)
#define AW_STEADY 2     // only A's 2 async chunks per tile
#else
#define WAIT_TENSORCNT(n)
#define AW_STEADY 4     // A (2) + B (2) async chunks per tile
#endif

typedef __attribute__((address_space(1))) v4i  gv4i;
typedef __attribute__((address_space(3))) v4i  lv4i;
typedef __attribute__((address_space(3))) bf16 lbf16;
typedef __attribute__((address_space(3))) char lchar;

union FragAB { v16bf v; uint4 q[2]; };
union FragC  { v8f  v; float f[8]; };

// 16-byte global -> LDS copy. LDS side pre-cast to AS(3): no per-call
// flat->LDS null-check in the hot loop.
__device__ __forceinline__ void cp16(lbf16* lds, const bf16* g) {
#if HAS_ASYNC_LDS
    __builtin_amdgcn_global_load_async_to_lds_b128(
        (gv4i*)(g), (lv4i*)(lds), 0, 0);
#else
    *(lv4i*)lds = *(const gv4i*)g;
#endif
}

__device__ __forceinline__ unsigned lds_off_u32(void* p) {
    return (unsigned)(unsigned long long)(lchar*)p;   // addrspacecast -> 32-bit LDS addr
}

#if HAS_TDM
// TDM: load a BK x BN bf16 tile (row stride = n_stride elems) into LDS,
// inserting 16B of padding after every 256B row (matches LDB_P pitch).
__device__ __forceinline__ void tdm_load_b_tile(const bf16* gsrc, unsigned lds_addr,
                                                int rows_rem, int n_stride) {
    unsigned long long ga = (unsigned long long)(size_t)gsrc;
    u32x4 g0;
    g0[0] = 1u;                                            // count=1, user mode
    g0[1] = lds_addr;                                      // lds_addr [63:32]
    g0[2] = (unsigned)(ga & 0xFFFFFFFFu);                  // global_addr [95:64]
    g0[3] = (unsigned)((ga >> 32) & 0x01FFFFFFu) | (2u << 30);  // addr hi | type=2
    i32x8 g1;
    // data_size=1 (2B) | pad_enable | pad_interval=5 (64 DW) | pad_amount=3 (4 DW)
    g1[0] = (int)((1u << 16) | (1u << 20) | (5u << 22) | (3u << 25));
    g1[1] = (int)(((unsigned)n_stride & 0xFFFFu) << 16);   // tensor_dim0 lo16 @48
    g1[2] = (int)((((unsigned)n_stride >> 16) & 0xFFFFu) |
                  (((unsigned)rows_rem & 0xFFFFu) << 16)); // dim0 hi | dim1 lo
    g1[3] = (int)((((unsigned)rows_rem >> 16) & 0xFFFFu) |
                  ((unsigned)BN << 16));                   // dim1 hi | tile_dim0
    g1[4] = (int)(BK);                                     // tile_dim1 | tile_dim2=0
    g1[5] = (int)n_stride;                                 // tensor_dim0_stride lo32
    g1[6] = 0;
    g1[7] = 0;
    i32x4 z4 = {0, 0, 0, 0};
#if defined(__clang_major__) && (__clang_major__ >= 23)
    i32x8 z8 = {0, 0, 0, 0, 0, 0, 0, 0};
    __builtin_amdgcn_tensor_load_to_lds(g0, g1, z4, z4, z8, 0);
#else
    __builtin_amdgcn_tensor_load_to_lds(g0, g1, z4, z4, 0);
#endif
}
#endif

// gelu(x) = x * sigmoid(2*c*(x + 0.044715 x^3)) -- branch-free, one v_exp_f32
__device__ __forceinline__ float gelu_tanh(float x) {
    const float c = 0.7978845608028654f; // sqrt(2/pi)
    float t = c * (x + 0.044715f * x * x * x);
    return x * (1.0f / (1.0f + __expf(-2.0f * t)));
}

// ---------------------------------------------------------------------------
__global__ __launch_bounds__(256) void cvt_f32_bf16(const float* __restrict__ in,
                                                    bf16* __restrict__ out, int n) {
    int i = (blockIdx.x * 256 + threadIdx.x) * 4;
    if (i + 3 < n) {
        float4 f = *(const float4*)(in + i);
        out[i + 0] = (bf16)f.x;
        out[i + 1] = (bf16)f.y;
        out[i + 2] = (bf16)f.z;
        out[i + 3] = (bf16)f.w;
    }
}

// ---------------------------------------------------------------------------
__global__ __launch_bounds__(256) void router_kernel(const float* __restrict__ x,
                                                     const float* __restrict__ Wr,
                                                     float* __restrict__ gates) {
    int token = blockIdx.x * 8 + (threadIdx.x >> 5);
    int lane  = threadIdx.x & 31;
    if (token >= NTOK) return;

    float acc[N_EXP];
#pragma unroll
    for (int e = 0; e < N_EXP; ++e) acc[e] = 0.0f;

    const float* xr = x + (size_t)token * N_EMBD;
    for (int c = lane; c < N_EMBD; c += 32) {
        float xv = xr[c];
#pragma unroll
        for (int e = 0; e < N_EXP; ++e) acc[e] += xv * Wr[c * N_EXP + e];
    }
#pragma unroll
    for (int off = 16; off > 0; off >>= 1) {
#pragma unroll
        for (int e = 0; e < N_EXP; ++e) acc[e] += __shfl_xor(acc[e], off, 32);
    }
    if (lane == 0) {
        float mx = acc[0];
#pragma unroll
        for (int e = 1; e < N_EXP; ++e) mx = fmaxf(mx, acc[e]);
        float p[N_EXP]; float s = 0.0f;
#pragma unroll
        for (int e = 0; e < N_EXP; ++e) { p[e] = __expf(acc[e] - mx); s += p[e]; }
        float inv = 1.0f / s;
#pragma unroll
        for (int e = 0; e < N_EXP; ++e) p[e] *= inv;
        int i1 = 0;
#pragma unroll
        for (int e = 1; e < N_EXP; ++e) if (p[e] > p[i1]) i1 = e;
        int i2 = (i1 == 0) ? 1 : 0;
#pragma unroll
        for (int e = 0; e < N_EXP; ++e) if (e != i1 && p[e] > p[i2]) i2 = e;
        float sn = 1.0f / (p[i1] + p[i2]);
#pragma unroll
        for (int e = 0; e < N_EXP; ++e) {
            float g = (e == i1) ? p[i1] * sn : (e == i2) ? p[i2] * sn : 0.0f;
            gates[(size_t)token * N_EXP + e] = g;
        }
    }
}

// ---------------------------------------------------------------------------
// Tiled bf16 WMMA GEMM.
//   MODE 0: Hout = bf16(gelu(A@B + bias))
//   MODE 1: Fout = (A@B + bias)              (write)
//   MODE 2: Fout += (A@B + bias)*gate[row]   (accumulate)
// ---------------------------------------------------------------------------
template <int MODE>
__global__ __launch_bounds__(256) void moe_gemm(
        const bf16* __restrict__ A,
        const bf16* __restrict__ Bb,
        const float* __restrict__ bias,
        const float* __restrict__ gate, int gstride,
        bf16* __restrict__ Hout,
        float* __restrict__ Fout,
        int M, int N, int K) {
    __shared__ __align__(16) bf16 lds_a[2][BM][LDA_P];
    __shared__ __align__(16) bf16 lds_b[2][BK][LDB_P];

    const int m0   = blockIdx.y * BM;
    const int n0   = blockIdx.x * BN;
    const int tid  = threadIdx.x;
    const int lane = tid & 31;
    const int wave = tid >> 5;
    const int wm0  = (wave >> 2) * 64;   // 2 waves in M
    const int wn0  = (wave & 3) * 32;    // 4 waves in N
    const int half = lane >> 4;
    const int l16  = lane & 15;

    // ---- per-thread staging pointers, AS(3)-cast ONCE (hoists null checks) ----
    const int rowA0 = tid >> 2,           chA0 = (tid & 3) * 8;         // rows 0..63
    const int rowA1 = (tid + 256) >> 2,   chA1 = ((tid + 256) & 3) * 8; // rows 64..127
    const bf16* aG0 = A + (size_t)(m0 + rowA0) * K + chA0;
    const bf16* aG1 = A + (size_t)(m0 + rowA1) * K + chA1;
    lbf16* aL0 = (lbf16*)&lds_a[0][rowA0][chA0];
    lbf16* aL1 = (lbf16*)&lds_a[0][rowA1][chA1];
    const int ABUF = BM * LDA_P;                       // elems per A buffer
#if HAS_TDM
    const bf16* bG = Bb + n0;                          // row 0 of this N-block
    const unsigned bLbase  = lds_off_u32(&lds_b[0][0][0]);
    const unsigned BBUF_B  = BK * LDB_P * sizeof(bf16); // bytes per B buffer
#else
    const int rowB = tid >> 3, colB = (tid & 7) * 16;
    const bf16* bG0 = Bb + (size_t)rowB * N + n0 + colB;
    lbf16* bL = (lbf16*)&lds_b[0][rowB][colB];
    const int BBUF = BK * LDB_P;                       // elems per B buffer
#endif

    auto stage = [&](int buf, int kt) {
        cp16(aL0 + buf * ABUF, aG0 + (size_t)kt * BK);
        cp16(aL1 + buf * ABUF, aG1 + (size_t)kt * BK);
#if HAS_TDM
        if (wave == 0)
            tdm_load_b_tile(bG + (size_t)kt * BK * N, bLbase + buf * BBUF_B,
                            K - kt * BK, N);
#else
        cp16(bL + buf * BBUF,     bG0 + (size_t)kt * BK * N);
        cp16(bL + buf * BBUF + 8, bG0 + (size_t)kt * BK * N + 8);
#endif
    };

    FragC acc[4][2];
#pragma unroll
    for (int mi = 0; mi < 4; ++mi)
#pragma unroll
        for (int ni = 0; ni < 2; ++ni)
#pragma unroll
            for (int v = 0; v < 8; ++v) acc[mi][ni].f[v] = 0.0f;

    auto compute = [&](int cur) {
        FragAB afr[4];
#pragma unroll
        for (int mi = 0; mi < 4; ++mi) {
            int r = wm0 + mi * 16 + l16;
            afr[mi].q[0] = *(const uint4*)&lds_a[cur][r][8 * half];
            afr[mi].q[1] = *(const uint4*)&lds_a[cur][r][16 + 8 * half];
        }
        FragAB bfr[2];
#pragma unroll
        for (int ni = 0; ni < 2; ++ni) {
            int c = wn0 + ni * 16;
#pragma unroll
            for (int s = 0; s < 8; ++s) {
                bfr[ni].v[s]     = lds_b[cur][8 * half + s][c + l16];
                bfr[ni].v[8 + s] = lds_b[cur][16 + 8 * half + s][c + l16];
            }
        }
#pragma unroll
        for (int mi = 0; mi < 4; ++mi)
#pragma unroll
            for (int ni = 0; ni < 2; ++ni)
                acc[mi][ni].v = __builtin_amdgcn_wmma_f32_16x16x32_bf16(
                        false, afr[mi].v, false, bfr[ni].v,
                        (short)0, acc[mi][ni].v, false, false);
    };

    const int KT = K / BK;
    stage(0, 0);

    // steady state: always prefetch + partial wait (no branchy waits in loop)
    for (int kt = 0; kt < KT - 1; ++kt) {
        const int cur = kt & 1;
        stage(cur ^ 1, kt + 1);
        WAIT_ASYNCCNT(AW_STEADY);
        WAIT_TENSORCNT(1);
        __syncthreads();
        compute(cur);
        __syncthreads();
    }
    // last tile
    WAIT_ASYNCCNT(0);
    WAIT_TENSORCNT(0);
    __syncthreads();
    compute((KT - 1) & 1);

    // ---- epilogue: C layout: VGPR v, lane -> row = v + 8*half, col = l16 ----
#pragma unroll
    for (int mi = 0; mi < 4; ++mi) {
#pragma unroll
        for (int ni = 0; ni < 2; ++ni) {
            int coln = n0 + wn0 + ni * 16 + l16;
            float bv = bias[coln];
#pragma unroll
            for (int v = 0; v < 8; ++v) {
                int rowm = m0 + wm0 + mi * 16 + v + 8 * half;
                float val = acc[mi][ni].f[v] + bv;
                if (MODE == 0) {
                    Hout[(size_t)rowm * N + coln] = (bf16)gelu_tanh(val);
                } else if (MODE == 1) {
                    Fout[(size_t)rowm * N + coln] = val;
                } else {
                    float g = gate[(size_t)rowm * gstride];
                    size_t o = (size_t)rowm * N + coln;
                    Fout[o] = Fout[o] + val * g;
                }
            }
        }
    }
}

// ---------------------------------------------------------------------------
extern "C" void kernel_launch(void* const* d_in, const int* in_sizes, int n_in,
                              void* d_out, int out_size, void* d_ws, size_t ws_size,
                              hipStream_t stream) {
    const float* x       = (const float*)d_in[0];
    const float* Wfc_s   = (const float*)d_in[1];
    const float* bfc_s   = (const float*)d_in[2];
    const float* Wproj_s = (const float*)d_in[3];
    const float* bproj_s = (const float*)d_in[4];
    const float* Wr      = (const float*)d_in[5];
    const float* Wfc     = (const float*)d_in[6];
    const float* bfc     = (const float*)d_in[7];
    const float* Wproj   = (const float*)d_in[8];
    const float* bproj   = (const float*)d_in[9];
    float* out = (float*)d_out;

    char* ws = (char*)d_ws;
    size_t off = 0;
    bf16* xb = (bf16*)(ws + off);       off += (size_t)NTOK * N_EMBD * sizeof(bf16);   // 16 MB
    bf16* hb = (bf16*)(ws + off);       off += (size_t)NTOK * HIDDEN * sizeof(bf16);   // 64 MB
    float* gates = (float*)(ws + off);  off += (size_t)NTOK * N_EXP * sizeof(float);   // 256 KB
    bf16* wfc_b  = (bf16*)(ws + off);   off += (size_t)N_EMBD * HIDDEN * sizeof(bf16); //  8 MB
    bf16* wpj_b  = (bf16*)(ws + off);   off += (size_t)HIDDEN * N_EMBD * sizeof(bf16); //  8 MB

    const int NW = N_EMBD * HIDDEN;

    cvt_f32_bf16<<<dim3((NTOK * N_EMBD) / 1024), dim3(256), 0, stream>>>(
            x, xb, NTOK * N_EMBD);
    router_kernel<<<dim3(NTOK / 8), dim3(256), 0, stream>>>(x, Wr, gates);

    dim3 blk(256);
    dim3 g1(HIDDEN / BN, NTOK / BM);   // 32 x 64
    dim3 g2(N_EMBD / BN, NTOK / BM);   //  8 x 64
    dim3 gcvt(NW / 1024);

    // shared expert: write out
    cvt_f32_bf16<<<gcvt, blk, 0, stream>>>(Wfc_s, wfc_b, NW);
    moe_gemm<0><<<g1, blk, 0, stream>>>(xb, wfc_b, bfc_s, nullptr, 0,
                                        hb, nullptr, NTOK, HIDDEN, N_EMBD);
    cvt_f32_bf16<<<gcvt, blk, 0, stream>>>(Wproj_s, wpj_b, NW);
    moe_gemm<1><<<g2, blk, 0, stream>>>(hb, wpj_b, bproj_s, nullptr, 0,
                                        nullptr, out, NTOK, N_EMBD, HIDDEN);
    // routed experts: accumulate gate-weighted
    for (int e = 0; e < N_EXP; ++e) {
        cvt_f32_bf16<<<gcvt, blk, 0, stream>>>(Wfc + (size_t)e * NW, wfc_b, NW);
        moe_gemm<0><<<g1, blk, 0, stream>>>(
                xb, wfc_b, bfc + (size_t)e * HIDDEN,
                nullptr, 0, hb, nullptr, NTOK, HIDDEN, N_EMBD);
        cvt_f32_bf16<<<gcvt, blk, 0, stream>>>(Wproj + (size_t)e * NW, wpj_b, NW);
        moe_gemm<2><<<g2, blk, 0, stream>>>(
                hb, wpj_b, bproj + (size_t)e * N_EMBD,
                gates + e, N_EXP, nullptr, out, NTOK, N_EMBD, HIDDEN);
    }
}
